// LearnedTripleConnect_70574902608415
// MI455X (gfx1250) — compile-verified, matrix operand order checked by reading, not run on hardware
//
#include <hip/hip_runtime.h>
#include <hip/hip_bf16.h>
#include <math.h>

// Problem constants (reference: B=8, N=8192, S=8, D_IN=64, D_OUT=64)
#define BB 8
#define NN 8192
#define SS 8
#define LDS_STRIDE 132   // 128 + 4 dword padding -> conflict-free strided access

typedef __attribute__((ext_vector_type(16))) _Float16 v16h;
typedef __attribute__((ext_vector_type(8)))  float    v8f;

// K index inside a 32-wide chunk for slot i (0..15) and lane-group g (0..1),
// per CDNA5 ISA 16-bit A/B fragment layout:
//   g=0: K = {0..7, 16..23}; g=1: K = {8..15, 24..31}
__device__ __forceinline__ int kmap(int i, int g) {
    return i + (i >= 8 ? 8 : 0) + 8 * g;
}

// ---------------------------------------------------------------------------
// Prep kernel: convert W1 (192x128) and W2 (128x64) f32 -> f16 and pack them
// into WMMA B-fragment order. Fragment (c,t) of W1 covers K rows [32c,32c+32)
// and N cols [16t,16t+16). Per-lane 16 halves stored contiguously (32B/lane).
//   W1: 6 K-chunks x 8 N-tiles = 48 frags  -> wp[0      .. 24576) halves
//   W2: 4 K-chunks x 4 N-tiles = 16 frags  -> wp[24576 .. 32768) halves
// ---------------------------------------------------------------------------
__global__ void pack_weights_kernel(const float* __restrict__ W1,
                                    const float* __restrict__ W2,
                                    _Float16* __restrict__ wp) {
    int tid = blockIdx.x * blockDim.x + threadIdx.x;   // 2048 units total
    if (tid < 48 * 32) {                                // W1 frags
        int frag = tid >> 5, lane = tid & 31;
        int c = frag >> 3, t = frag & 7;
        int g = lane >> 4, nn = lane & 15;
        _Float16* dst = wp + (size_t)tid * 16;
#pragma unroll
        for (int i = 0; i < 16; ++i) {
            int K = 32 * c + kmap(i, g);
            dst[i] = (_Float16)W1[K * 128 + 16 * t + nn];
        }
    } else if (tid < 48 * 32 + 16 * 32) {               // W2 frags
        int u = tid - 48 * 32;
        int frag = u >> 5, lane = u & 31;
        int c = frag >> 2, t = frag & 3;
        int g = lane >> 4, nn = lane & 15;
        _Float16* dst = wp + 24576 + (size_t)u * 16;
#pragma unroll
        for (int i = 0; i < 16; ++i) {
            int K = 32 * c + kmap(i, g);
            dst[i] = (_Float16)W2[K * 64 + 16 * t + nn];
        }
    }
}

// Build one A fragment (16x32 f16) for lane (g,m) from a row of 64 f32.
// localChunk selects K range [32*lc, 32*lc+32). Two contiguous 8-float runs.
__device__ __forceinline__ v16h load_a_chunk(const float* __restrict__ row,
                                             int g, int lc) {
    const float* p0 = row + lc * 32 + 8 * g;
    const float* p1 = p0 + 16;
    v16h r;
#pragma unroll
    for (int q = 0; q < 8; ++q) r[q] = (_Float16)p0[q];
#pragma unroll
    for (int q = 0; q < 8; ++q) r[8 + q] = (_Float16)p1[q];
    return r;
}

// ---------------------------------------------------------------------------
// Main kernel: block = 16 (b,n) rows x 8 waves (one s per wave), 256 threads.
// ---------------------------------------------------------------------------
__global__ __launch_bounds__(256, 2)
void LearnedTripleConnect_70574902608415_kernel(
        const float* __restrict__ x,       // (B,N,64)
        const int*   __restrict__ j_idx,   // (B,N,S)
        const int*   __restrict__ k_idx,   // (B,N,S)
        const float* __restrict__ b1,      // (128)
        const float* __restrict__ b2,      // (64)
        const _Float16* __restrict__ wp,   // packed W1 | W2 fragments
        float* __restrict__ out) {         // (B,N,64)
    __shared__ float hsum[16 * LDS_STRIDE];

    const int tid  = threadIdx.x;
    const int wave = tid >> 5;          // 0..7  == s
    const int lane = tid & 31;
    const int g    = lane >> 4;         // lane group (K-half selector)
    const int m    = lane & 15;         // row within tile / col within tile

    const int bn_base = blockIdx.x * 16;

    // zero the cross-wave h accumulator
    for (int i = tid; i < 16 * LDS_STRIDE; i += 256) hsum[i] = 0.0f;
    __syncthreads();

    // --- gather the three 64-vectors for row (bn_base+m, s=wave) ------------
    const int bn = bn_base + m;
    const int b  = bn >> 13;            // N = 8192
    const float* rowi = x + (size_t)bn * 64;
    const int j = j_idx[bn * SS + wave];
    const int k = k_idx[bn * SS + wave];
    const float* rowj = x + ((size_t)(b << 13) + j) * 64;
    const float* rowk = x + ((size_t)(b << 13) + k) * 64;

    // A tile 16x192 as 6 fragments, already in WMMA register layout
    v16h a[6];
    a[0] = load_a_chunk(rowi, g, 0);
    a[1] = load_a_chunk(rowi, g, 1);
    a[2] = load_a_chunk(rowj, g, 0);
    a[3] = load_a_chunk(rowj, g, 1);
    a[4] = load_a_chunk(rowk, g, 0);
    a[5] = load_a_chunk(rowk, g, 1);

    // --- matmul1: (16x192)@(192x128), bias, exact GELU, reduce over s -------
    const v16h* w1f = (const v16h*)wp;                 // 48 frags x 32 lanes
#pragma unroll
    for (int t = 0; t < 8; ++t) {                      // N-tiles of W1
        v8f acc = {};
#pragma unroll
        for (int c = 0; c < 6; ++c) {                  // K-chunks
            v16h wf = w1f[(c * 8 + t) * 32 + lane];
            acc = __builtin_amdgcn_wmma_f32_16x16x32_f16(
                      false, a[c], false, wf, (short)0, acc, false, false);
        }
        const float bias = b1[t * 16 + m];             // col = t*16+m, all rows
#pragma unroll
        for (int r = 0; r < 8; ++r) {                  // row M = r + 8*g
            float v = acc[r] + bias;
            float h = 0.5f * v * (1.0f + erff(v * 0.70710678118654752f));
            atomicAdd(&hsum[(r + 8 * g) * LDS_STRIDE + t * 16 + m], h);
        }
    }
    __syncthreads();

    // --- matmul2: mean_s(h) @ W2 + b2  (waves 0..3, one 16-col tile each) ---
    if (wave < 4) {
        const int t2 = wave;
        const v16h* w2f = (const v16h*)(wp + 24576);
        v8f acc = {};
#pragma unroll
        for (int c = 0; c < 4; ++c) {                  // K-chunks of 32
            const float* p0 = &hsum[m * LDS_STRIDE + c * 32 + 8 * g];
            v16h af;
#pragma unroll
            for (int q = 0; q < 8; ++q) af[q] = (_Float16)p0[q];
#pragma unroll
            for (int q = 0; q < 8; ++q) af[8 + q] = (_Float16)p0[16 + q];
            v16h wf = w2f[(c * 4 + t2) * 32 + lane];
            acc = __builtin_amdgcn_wmma_f32_16x16x32_f16(
                      false, af, false, wf, (short)0, acc, false, false);
        }
        const float bias = b2[t2 * 16 + m];
#pragma unroll
        for (int r = 0; r < 8; ++r) {                  // row M = r + 8*g
            out[(size_t)(bn_base + r + 8 * g) * 64 + t2 * 16 + m] =
                acc[r] * 0.125f + bias;                // mean over S = 8
        }
    }
}

extern "C" void kernel_launch(void* const* d_in, const int* in_sizes, int n_in,
                              void* d_out, int out_size, void* d_ws, size_t ws_size,
                              hipStream_t stream) {
    const float* x     = (const float*)d_in[0];
    const int*   j_idx = (const int*)  d_in[1];
    const int*   k_idx = (const int*)  d_in[2];
    const float* W1    = (const float*)d_in[3];
    const float* b1    = (const float*)d_in[4];
    const float* W2    = (const float*)d_in[5];
    const float* b2    = (const float*)d_in[6];
    float* out = (float*)d_out;
    _Float16* wp = (_Float16*)d_ws;    // 64 KB used: 48KB W1 + 16KB W2 packed

    // Pack weights into WMMA fragment layout (2048 pack units)
    pack_weights_kernel<<<8, 256, 0, stream>>>(W1, W2, wp);

    // B*N / 16 rows-per-block = 65536/16 = 4096 blocks, 8 waves each
    LearnedTripleConnect_70574902608415_kernel<<<4096, 256, 0, stream>>>(
        x, j_idx, k_idx, b1, b2, wp, out);
}
